// MultiModalFusion_16011638080125
// MI455X (gfx1250) — matrix operand order repaired and checked by reading, not compiled, not run on hardware
//
#include <hip/hip_runtime.h>

// ---------------------------------------------------------------------------
// MultiModalFusion for MI455X (gfx1250), wave32 + WMMA.
//
// Algebraic simplification of the reference:
//   fused_feat == 0.5*(feat0+feat1)   (softmax over identical modality axis)
//   auto_enc_loss needs the real conv autoencoder -> bf16 WMMA implicit GEMM.
//   The SVD/argmax path has no effect on either output and is skipped.
//
// Implicit-GEMM conv inner loop: all A and B fragments are contiguous
// ds_load_b128 pairs (K-order k = tap*CI+ci with channel-last LDS tile makes
// every 16-aligned K-run linear in LDS).
// ---------------------------------------------------------------------------

typedef __bf16 bf16_t;
typedef __attribute__((ext_vector_type(16))) __bf16 v16bf;
typedef __attribute__((ext_vector_type(8)))  float  v8f;
typedef __attribute__((ext_vector_type(4)))  unsigned int u32x4;

__device__ __forceinline__ unsigned short f2bf(float f) {
    union { float f; unsigned int u; } c; c.f = f;
    unsigned int u = c.u;
    unsigned int r = u + 0x7FFFu + ((u >> 16) & 1u);   // round-to-nearest-even
    return (unsigned short)(r >> 16);
}

union Frag16 { v16bf v; u32x4 q[2]; };

// ---------------------------------------------------------------------------
// Implicit-GEMM 3x3 SAME conv + bias + ReLU, bf16 WMMA, f32 accumulate.
//   in : NCHW, float (FIRST) or raw-bf16 ushort (!FIRST)
//   out: NCHW raw-bf16 ushort (when !LAST)
//   LAST: instead of storing, accumulates sum((ref - y)^2) into partial[blk].
// ---------------------------------------------------------------------------
template<int CI, int CO, int TILEW, bool FIRST, bool LAST>
__global__ __launch_bounds__((TILEW / 16) * 32)
void conv3x3_wmma_kernel(const void* __restrict__ in,
                         const float* __restrict__ wgt,
                         const float* __restrict__ bias,
                         unsigned short* __restrict__ out,
                         const float* __restrict__ ref,
                         float* __restrict__ partial,
                         int H, int W)
{
    constexpr int KTOT = CI * 9;
    constexpr int KPAD = (KTOT + 31) & ~31;
    constexpr int COLS = TILEW + 2;
    constexpr int MT   = CO / 16;       // 16-row output-channel tiles
    constexpr int NW   = TILEW / 16;    // waves per block

    extern __shared__ char smem[];
    unsigned short* sw = (unsigned short*)smem;                      // [CO][KPAD]
    unsigned short* si = (unsigned short*)(smem + CO * KPAD * 2);    // [3][COLS][CI]

    const int tid  = threadIdx.x;
    const int lane = tid & 31;
    const int wave = tid >> 5;
    const int nthr = NW * 32;

    const int b  = blockIdx.z;
    const int h  = blockIdx.y;
    const int w0 = blockIdx.x * TILEW;

    // ---- stage weights into LDS as bf16, k = tap*CI + ci, zero-pad to KPAD
    for (int e = tid; e < CO * KPAD; e += nthr) {
        int co = e / KPAD;
        int k  = e - co * KPAD;
        unsigned short v = 0;
        if (k < KTOT) {
            int tap = k / CI;            // CI is a power of two -> shift
            int ci  = k & (CI - 1);
            v = f2bf(wgt[(co * CI + ci) * 9 + tap]);
        }
        sw[e] = v;
    }
    // ---- stage haloed input tile [3][COLS][CI] as bf16 (zeros outside image)
    for (int e = tid; e < 3 * COLS * CI; e += nthr) {
        int x  = e % COLS;
        int t2 = e / COLS;
        int y  = t2 % 3;
        int ci = t2 / 3;
        int gy = h - 1 + y;
        int gx = w0 - 1 + x;
        unsigned short v = 0;
        if (gy >= 0 && gy < H && gx >= 0 && gx < W) {
            size_t gidx = ((size_t)(b * CI + ci) * H + gy) * W + gx;
            if constexpr (FIRST) v = f2bf(((const float*)in)[gidx]);
            else                 v = ((const unsigned short*)in)[gidx];
        }
        si[(y * COLS + x) * CI + ci] = v;
    }
    __syncthreads();

    const int pxl = wave * 16 + (lane & 15);   // pixel within tile (N index)
    const int hi  = lane >> 4;                 // lane half

    v8f acc[MT];
    #pragma unroll
    for (int mt = 0; mt < MT; ++mt)
        #pragma unroll
        for (int r = 0; r < 8; ++r) acc[mt][r] = 0.f;

    // base LDS offset (in shorts) of this lane's pixel column
    const int pxbase = pxl * CI;

    #pragma unroll
    for (int ks = 0; ks < KPAD; ks += 32) {
        // ---- B fragment (32x16 KxN): half0 lanes K=ks..ks+15, half1 ks+16..31
        // Contiguous in LDS: addr(k) = (dy*COLS+px)*CI + (k - dy*3*CI).
        Frag16 bf;
        const int kb = ks + (hi << 4);
        if (KPAD == KTOT || kb < KTOT) {         // folds away unless padded
            const int dy = kb / (3 * CI);        // constant-ish: row of kernel
            const unsigned short* bbase =
                si + (dy * COLS) * CI + pxbase + (kb - dy * 3 * CI);
            bf.q[0] = *(const u32x4*)(bbase);        // K = kb   .. kb+7
            bf.q[1] = *(const u32x4*)(bbase + 8);    // K = kb+8 .. kb+15
        } else {
            #pragma unroll
            for (int j = 0; j < 4; ++j) { bf.q[0][j] = 0u; bf.q[1][j] = 0u; }
        }
        // ---- A fragments: lanes0-15 K{+0..15}, lanes16-31 {+16..31}
        const int ko = ks + (hi << 4);
        #pragma unroll
        for (int mt = 0; mt < MT; ++mt) {
            Frag16 af;
            const unsigned short* abase = sw + ((mt * 16 + (lane & 15)) * KPAD + ko);
            af.q[0] = *(const u32x4*)(abase);        // K = ko    .. ko+7
            af.q[1] = *(const u32x4*)(abase + 8);    // K = ko+8  .. ko+15
            acc[mt] = __builtin_amdgcn_wmma_f32_16x16x32_bf16(
                false, af.v, false, bf.v, (short)0, acc[mt], false, false);
        }
    }

    // ---- epilogue: bias + ReLU, then store bf16 or accumulate squared error
    const int gx = w0 + pxl;
    float lsum = 0.f;
    #pragma unroll
    for (int mt = 0; mt < MT; ++mt) {
        #pragma unroll
        for (int r = 0; r < 8; ++r) {
            int   co = mt * 16 + (hi << 3) + r;   // C/D layout: VGPR r -> M=r(+8)
            float v  = acc[mt][r] + bias[co];
            v = v > 0.f ? v : 0.f;
            size_t oidx = ((size_t)(b * CO + co) * H + h) * W + gx;
            if constexpr (LAST) {
                float e = ref[oidx] - v;
                lsum += e * e;
            } else {
                out[oidx] = f2bf(v);
            }
        }
    }
    if constexpr (LAST) {
        #pragma unroll
        for (int off = 16; off > 0; off >>= 1)
            lsum += __shfl_down(lsum, off, 32);
        __syncthreads();                 // all LDS reads done; reuse smem
        float* red = (float*)smem;
        if (lane == 0) red[wave] = lsum;
        __syncthreads();
        if (tid == 0) {
            float s = 0.f;
            for (int i = 0; i < NW; ++i) s += red[i];
            partial[((size_t)blockIdx.z * gridDim.y + blockIdx.y) * gridDim.x
                    + blockIdx.x] = s;
        }
    }
}

// ---------------------------------------------------------------------------
// fused_feat = 0.5*(feat0 + feat1), float4-vectorized (pure-bandwidth pass)
// ---------------------------------------------------------------------------
__global__ void fuse_avg_kernel(const float4* __restrict__ a,
                                const float4* __restrict__ b,
                                float4* __restrict__ o, int n4)
{
    int i = blockIdx.x * blockDim.x + threadIdx.x;
    if (i < n4) {
        float4 x = a[i], y = b[i];
        o[i] = make_float4(0.5f * (x.x + y.x), 0.5f * (x.y + y.y),
                           0.5f * (x.z + y.z), 0.5f * (x.w + y.w));
    }
}

// ---------------------------------------------------------------------------
// Deterministic final reduction of per-block loss partials.
// ---------------------------------------------------------------------------
__global__ void loss_reduce_kernel(const float* __restrict__ partials, int n,
                                   float* __restrict__ outLoss, float invN)
{
    __shared__ float s[256];
    float acc = 0.f;
    for (int i = threadIdx.x; i < n; i += 256) acc += partials[i];
    s[threadIdx.x] = acc;
    __syncthreads();
    for (int o = 128; o > 0; o >>= 1) {
        if (threadIdx.x < o) s[threadIdx.x] += s[threadIdx.x + o];
        __syncthreads();
    }
    if (threadIdx.x == 0) outLoss[0] = s[0] * invN;
}

// ---------------------------------------------------------------------------
extern "C" void kernel_launch(void* const* d_in, const int* in_sizes, int n_in,
                              void* d_out, int out_size, void* d_ws, size_t ws_size,
                              hipStream_t stream)
{
    const float* feat0 = (const float*)d_in[0];
    const float* feat1 = (const float*)d_in[1];
    const float* w1 = (const float*)d_in[2];
    const float* b1 = (const float*)d_in[3];
    const float* w2 = (const float*)d_in[4];
    const float* b2 = (const float*)d_in[5];
    const float* w3 = (const float*)d_in[6];
    const float* b3 = (const float*)d_in[7];
    const float* w4 = (const float*)d_in[8];
    const float* b4 = (const float*)d_in[9];
    // d_in[10]: modality_adapter (unused by the reference forward)

    const int B = 8, C = 64, H = 128, W = 256;
    const size_t HW  = (size_t)H * W;
    const size_t NEL = (size_t)B * C * HW;           // 16,777,216

    float* outFused = (float*)d_out;
    float* outLoss  = (float*)d_out + NEL;

    // workspace: act1 (32ch bf16, reused as act2) | mid (16ch bf16) | partials
    char* ws = (char*)d_ws;
    unsigned short* act1 = (unsigned short*)ws;
    unsigned short* mid  = (unsigned short*)(ws + (size_t)B * 32 * HW * 2);
    unsigned short* act2 = act1;  // act1 dead after layer 2 -> reuse
    float* partials = (float*)(ws + (size_t)B * 32 * HW * 2 + (size_t)B * 16 * HW * 2);

    auto smemB = [](int CI, int CO, int TILEW) -> size_t {
        int KTOT = CI * 9;
        int KPAD = (KTOT + 31) & ~31;
        return (size_t)(CO * KPAD + 3 * (TILEW + 2) * CI) * 2;
    };
    const size_t sm1 = smemB(64, 32, 64);    // 62,208 B
    const size_t sm2 = smemB(32, 16, 128);   // 34,176 B
    const size_t sm3 = smemB(16, 32, 128);   // 22,720 B
    const size_t sm4 = smemB(32, 64, 128);   // 61,824 B

    const dim3 g64(W / 64, H, B);    // layer 1 (4 waves/block)
    const dim3 g128(W / 128, H, B);  // layers 2-4 (8 waves/block)
    const int PB = B * H * (W / 128);          // per-feat partial count = 2048

    const float* feats[2] = { feat0, feat1 };
    for (int f = 0; f < 2; ++f) {
        const float* ff = feats[f];
        conv3x3_wmma_kernel<64, 32,  64, true,  false>
            <<<g64, 128, sm1, stream>>>(ff,   w1, b1, act1, nullptr, nullptr, H, W);
        conv3x3_wmma_kernel<32, 16, 128, false, false>
            <<<g128, 256, sm2, stream>>>(act1, w2, b2, mid,  nullptr, nullptr, H, W);
        conv3x3_wmma_kernel<16, 32, 128, false, false>
            <<<g128, 256, sm3, stream>>>(mid,  w3, b3, act2, nullptr, nullptr, H, W);
        conv3x3_wmma_kernel<32, 64, 128, false, true>
            <<<g128, 256, sm4, stream>>>(act2, w4, b4, nullptr, ff,
                                         partials + (size_t)f * PB, H, W);
    }

    loss_reduce_kernel<<<1, 256, 0, stream>>>(partials, 2 * PB, outLoss,
                                              1.0f / (float)NEL);

    const int n4 = (int)(NEL / 4);
    fuse_avg_kernel<<<(n4 + 255) / 256, 256, 0, stream>>>(
        (const float4*)feat0, (const float4*)feat1, (float4*)outFused, n4);
}
